// GCNLinkPredictor_76398878261367
// MI455X (gfx1250) — matrix operand order repaired and checked by reading, not compiled, not run on hardware
//
#include <hip/hip_runtime.h>
#include <hip/hip_bf16.h>
#include <stdint.h>

// ---------------------------------------------------------------------------
// GCN link predictor for MI455X (gfx1250, wave32, WMMA).
//   deg/dinv -> bf16-WMMA GEMM (x@W) -> self-loop+bias init -> edge scatter
//   (x2 layers, x2 graphs) -> fold Wproj@Wl1 -> per-node hp/hs GEMMs ->
//   per-edge gather + ReLU-dot link kernel.
// GEMM v2: one wave owns a 16-row tile and all 64 output columns
// (4 accumulators, 4 WMMAs/K-step) -> A streamed exactly once, non-temporal.
// ---------------------------------------------------------------------------

typedef __bf16 bf16_t;
typedef bf16_t v16bf __attribute__((ext_vector_type(16)));
typedef float  v8f   __attribute__((ext_vector_type(8)));
typedef float  f32x4 __attribute__((ext_vector_type(4)));

#define H 64
#define NP 100000
#define PD 1024
#define NS 50000
#define SD 512
#define EP 3200000
#define ES 1600000
#define EL 2000000

__device__ __forceinline__ bf16_t f2bf(float f) {
  return (bf16_t)f;   // fptrunc f32->bf16, RNE; native cvt on gfx1250
}

__device__ __forceinline__ void atomAddF(float* p, float v) {
  __hip_atomic_fetch_add(p, v, __ATOMIC_RELAXED, __HIP_MEMORY_SCOPE_AGENT);
}

// ----------------------------- degree kernels ------------------------------
__global__ void deg_fill(float* __restrict__ deg, int n) {
  int t = blockIdx.x * blockDim.x + threadIdx.x;
  if (t < n) deg[t] = 1.0f;                       // self-loop
}

__global__ void deg_edges(const int* __restrict__ dst, float* __restrict__ deg, int E) {
  int t = blockIdx.x * blockDim.x + threadIdx.x;
  if (t < E) atomAddF(deg + dst[t], 1.0f);
}

__global__ void deg_rsqrt(float* __restrict__ deg, int n) {
  int t = blockIdx.x * blockDim.x + threadIdx.x;
  if (t < n) deg[t] = rsqrtf(deg[t]);             // in-place -> dinv
}

// --------------------------- weight repack (B frag) ------------------------
// Wpacked[((ct*KT + kt)*32 + lane)*16 + i] = bf16(W[(kt*32 + (i<8 ? lh*8+i : 16+lh*8+i-8))*64 + ct*16 + lr])
// B-matrix 32x16 bf16 layout: lane half lh selects K {lh*8..+7} U {16+lh*8..+7}, col = lane&15.
__global__ void repack_w(const float* __restrict__ W, bf16_t* __restrict__ Wp, int K) {
  int KT = K >> 5;
  int tid = blockIdx.x * blockDim.x + threadIdx.x;
  int total = 4 * KT * 512;
  if (tid >= total) return;
  int i    = tid & 15;
  int lane = (tid >> 4) & 31;
  int kt   = (tid >> 9) % KT;
  int ct   = (tid >> 9) / KT;
  int lh = lane >> 4, lr = lane & 15;
  int k = kt * 32 + (i < 8 ? lh * 8 + i : 16 + lh * 8 + (i - 8));
  int n = ct * 16 + lr;
  Wp[tid] = f2bf(W[(size_t)k * 64 + n]);
}

// ------------------------------ WMMA GEMM ----------------------------------
// C[rows x 64] = A[rows x K] (fp32, converted inline to bf16) @ Wpacked.
// 128 threads = 4 waves; each wave owns ONE 16-row tile and all 4 column
// tiles (4 WMMAs per K-step). A is loaded once, non-temporally.
__global__ void __launch_bounds__(128)
gemm_bf16(const float* __restrict__ A, const bf16_t* __restrict__ Wp,
          float* __restrict__ C, int K, int rowTiles) {
  const int lane = threadIdx.x & 31;
  const int wave = threadIdx.x >> 5;
  const int tile = blockIdx.x * 4 + wave;
  if (tile >= rowTiles) return;                  // whole-wave exit, EXEC stays ~0
  const int lh = lane >> 4;
  const int lr = lane & 15;
  const int KT = K >> 5;
  const size_t cstride = (size_t)KT * 512;       // elements between column tiles
  const float* arow = A + (size_t)(tile * 16 + lr) * K;

  v8f acc[4];
#pragma unroll
  for (int ct = 0; ct < 4; ++ct)
    acc[ct] = (v8f){0.f, 0.f, 0.f, 0.f, 0.f, 0.f, 0.f, 0.f};

#pragma unroll 2
  for (int kt = 0; kt < KT; ++kt) {
    // A 16x32 bf16 fragment: lane half lh -> K in {lh*8..+7} U {16+lh*8..+7}
    f32x4 a0 = __builtin_nontemporal_load((const f32x4*)(arow + kt * 32 + lh * 8));
    f32x4 a1 = __builtin_nontemporal_load((const f32x4*)(arow + kt * 32 + lh * 8 + 4));
    f32x4 b0 = __builtin_nontemporal_load((const f32x4*)(arow + kt * 32 + 16 + lh * 8));
    f32x4 b1 = __builtin_nontemporal_load((const f32x4*)(arow + kt * 32 + 16 + lh * 8 + 4));
    v16bf afrag;
    afrag[0]  = f2bf(a0.x); afrag[1]  = f2bf(a0.y);
    afrag[2]  = f2bf(a0.z); afrag[3]  = f2bf(a0.w);
    afrag[4]  = f2bf(a1.x); afrag[5]  = f2bf(a1.y);
    afrag[6]  = f2bf(a1.z); afrag[7]  = f2bf(a1.w);
    afrag[8]  = f2bf(b0.x); afrag[9]  = f2bf(b0.y);
    afrag[10] = f2bf(b0.z); afrag[11] = f2bf(b0.w);
    afrag[12] = f2bf(b1.x); afrag[13] = f2bf(b1.y);
    afrag[14] = f2bf(b1.z); afrag[15] = f2bf(b1.w);

    const bf16_t* wk = Wp + (size_t)kt * 512 + (size_t)lane * 16;
#pragma unroll
    for (int ct = 0; ct < 4; ++ct) {
      v16bf bfrag = *(const v16bf*)(wk + (size_t)ct * cstride);
      acc[ct] = __builtin_amdgcn_wmma_f32_16x16x32_bf16(
          false, afrag, false, bfrag, (short)0, acc[ct], false, false);
    }
  }

  // C/D f32 16x16: VGPR r -> row lh*8+r, col = lr (+16*ct)
  float* crow = C + (size_t)(tile * 16 + lh * 8) * 64 + lr;
#pragma unroll
  for (int ct = 0; ct < 4; ++ct)
#pragma unroll
    for (int r = 0; r < 8; ++r)
      crow[(size_t)r * 64 + ct * 16] = acc[ct][r];
}

// --------------------------- GCN aggregation -------------------------------
// out[i] = h[i]*dinv[i]^2 + bias   (self-loop contribution + bias)
__global__ void scatter_init(const float* __restrict__ h, const float* __restrict__ dinv,
                             const float* __restrict__ bias, float* __restrict__ out, int N) {
  int tid = blockIdx.x * blockDim.x + threadIdx.x;
  if (tid >= N * H) return;
  int i = tid >> 6, f = tid & 63;
  float d = dinv[i];
  out[tid] = h[tid] * d * d + bias[f];
}

// one wave32 per edge: lanes cover 64 features as float2
__global__ void __launch_bounds__(256)
scatter_edges(const float* __restrict__ h, const int* __restrict__ ei,
              const float* __restrict__ dinv, float* __restrict__ out, int E) {
  int wave = threadIdx.x >> 5, lane = threadIdx.x & 31;
  int e = blockIdx.x * 8 + wave;
  if (e >= E) return;
  int src = ei[e];
  int dst = ei[(size_t)E + e];
  float norm = dinv[src] * dinv[dst];
  float2 v = ((const float2*)(h + (size_t)src * H))[lane];
  float* o = out + (size_t)dst * H + lane * 2;
  atomAddF(o,     v.x * norm);
  atomAddF(o + 1, v.y * norm);
}

// ----------------------- fold Wproj@Wl1 (tiny GEMM) ------------------------
__global__ void fold_weights(const float* __restrict__ Wproj, const float* __restrict__ bproj,
                             const float* __restrict__ Wl1, const float* __restrict__ bl1,
                             float* __restrict__ W1fold, float* __restrict__ b1fold) {
  int tid = blockIdx.x * blockDim.x + threadIdx.x;
  if (tid < 128 * 64) {
    int i = tid >> 6, n = tid & 63;
    float s = 0.f;
    for (int j = 0; j < 128; ++j) s += Wproj[i * 128 + j] * Wl1[j * 64 + n];
    W1fold[tid] = s;
  } else if (tid < 128 * 64 + 64) {
    int n = tid - 128 * 64;
    float s = bl1[n];
    for (int j = 0; j < 128; ++j) s += bproj[j] * Wl1[j * 64 + n];
    b1fold[n] = s;
  }
}

// ------------------------------ link kernel --------------------------------
// out[e] = sum_n relu(hp[ep[e]][n] + hs[es[e]][n] + b1fold[n]) * Wl2[n] + bl2
__global__ void __launch_bounds__(256)
link_pred(const float* __restrict__ hp, const float* __restrict__ hs,
          const int* __restrict__ ep, const int* __restrict__ es,
          const float* __restrict__ b1fold, const float* __restrict__ Wl2,
          const float* __restrict__ bl2, float* __restrict__ out, int E) {
  int wave = threadIdx.x >> 5, lane = threadIdx.x & 31;
  int e = blockIdx.x * 8 + wave;
  if (e >= E) return;
  int ip = ep[e], iq = es[e];
  float2 p = ((const float2*)(hp + (size_t)ip * H))[lane];
  float2 s = ((const float2*)(hs + (size_t)iq * H))[lane];
  float2 b = ((const float2*)b1fold)[lane];
  float2 w = ((const float2*)Wl2)[lane];
  float t = fmaxf(p.x + s.x + b.x, 0.f) * w.x
          + fmaxf(p.y + s.y + b.y, 0.f) * w.y;
#pragma unroll
  for (int off = 16; off > 0; off >>= 1) t += __shfl_xor(t, off, 32);
  if (lane == 0) out[e] = t + bl2[0];
}

// ---------------------------------------------------------------------------
extern "C" void kernel_launch(void* const* d_in, const int* in_sizes, int n_in,
                              void* d_out, int out_size, void* d_ws, size_t ws_size,
                              hipStream_t stream) {
  (void)in_sizes; (void)n_in; (void)out_size; (void)ws_size;
  const float* x_p   = (const float*)d_in[0];
  const float* x_s   = (const float*)d_in[1];
  const int*   ei_p  = (const int*)d_in[2];
  const int*   ei_s  = (const int*)d_in[3];
  const int*   ed_p  = (const int*)d_in[4];
  const int*   ed_s  = (const int*)d_in[5];
  const float* Wp1   = (const float*)d_in[6];
  const float* bp1   = (const float*)d_in[7];
  const float* Ws1   = (const float*)d_in[8];
  const float* bs1   = (const float*)d_in[9];
  const float* Wp2   = (const float*)d_in[10];
  const float* bp2   = (const float*)d_in[11];
  const float* Ws2   = (const float*)d_in[12];
  const float* bs2   = (const float*)d_in[13];
  const float* Wproj = (const float*)d_in[14];
  const float* bproj = (const float*)d_in[15];
  const float* Wl1   = (const float*)d_in[16];
  const float* bl1   = (const float*)d_in[17];
  const float* Wl2   = (const float*)d_in[18];
  const float* bl2   = (const float*)d_in[19];
  float* out = (float*)d_out;

  // bump allocator over workspace
  size_t off = 0;
  auto alloc = [&](size_t bytes) -> void* {
    void* p = (char*)d_ws + off;
    off += (bytes + 255) & ~(size_t)255;
    return p;
  };
  float* degp   = (float*)alloc((size_t)NP * 4);
  float* degs   = (float*)alloc((size_t)NS * 4);
  float* bufP1  = (float*)alloc((size_t)NP * H * 4);
  float* bufP2  = (float*)alloc((size_t)NP * H * 4);
  float* bufS1  = (float*)alloc((size_t)NS * H * 4);
  float* bufS2  = (float*)alloc((size_t)NS * H * 4);
  float* W1fold = (float*)alloc(128 * 64 * 4);
  float* b1fold = (float*)alloc(64 * 4);
  bf16_t* Wp1p  = (bf16_t*)alloc((size_t)4 * (PD / 32) * 512 * 2);
  bf16_t* Ws1p  = (bf16_t*)alloc((size_t)4 * (SD / 32) * 512 * 2);
  bf16_t* Wp2p  = (bf16_t*)alloc((size_t)4 * (H  / 32) * 512 * 2);
  bf16_t* Ws2p  = (bf16_t*)alloc((size_t)4 * (H  / 32) * 512 * 2);
  bf16_t* Wftop = (bf16_t*)alloc((size_t)4 * (H  / 32) * 512 * 2);
  bf16_t* Wfbot = (bf16_t*)alloc((size_t)4 * (H  / 32) * 512 * 2);

  auto cdiv = [](int a, int b) { return (a + b - 1) / b; };
  const int tilesP = NP / 16, tilesS = NS / 16;

  // 1) degrees + dinv
  deg_fill <<<cdiv(NP, 256), 256, 0, stream>>>(degp, NP);
  deg_fill <<<cdiv(NS, 256), 256, 0, stream>>>(degs, NS);
  deg_edges<<<cdiv(EP, 256), 256, 0, stream>>>(ei_p + EP, degp, EP);
  deg_edges<<<cdiv(ES, 256), 256, 0, stream>>>(ei_s + ES, degs, ES);
  deg_rsqrt<<<cdiv(NP, 256), 256, 0, stream>>>(degp, NP);
  deg_rsqrt<<<cdiv(NS, 256), 256, 0, stream>>>(degs, NS);

  // 2) repack conv weights to WMMA B-fragment layout (bf16)
  repack_w<<<cdiv(4 * (PD / 32) * 512, 256), 256, 0, stream>>>(Wp1, Wp1p, PD);
  repack_w<<<cdiv(4 * (SD / 32) * 512, 256), 256, 0, stream>>>(Ws1, Ws1p, SD);
  repack_w<<<cdiv(4 * (H  / 32) * 512, 256), 256, 0, stream>>>(Wp2, Wp2p, H);
  repack_w<<<cdiv(4 * (H  / 32) * 512, 256), 256, 0, stream>>>(Ws2, Ws2p, H);

  // 3) fold linear layers of the link MLP, repack halves
  fold_weights<<<cdiv(128 * 64 + 64, 256), 256, 0, stream>>>(Wproj, bproj, Wl1, bl1, W1fold, b1fold);
  repack_w<<<cdiv(4 * (H / 32) * 512, 256), 256, 0, stream>>>(W1fold,           Wftop, H);
  repack_w<<<cdiv(4 * (H / 32) * 512, 256), 256, 0, stream>>>(W1fold + 64 * 64, Wfbot, H);

  // 4) protein GCN: conv1, conv2, then per-node link projection
  gemm_bf16    <<<cdiv(tilesP, 4), 128, 0, stream>>>(x_p, Wp1p, bufP1, PD, tilesP);
  scatter_init <<<cdiv(NP * H, 256), 256, 0, stream>>>(bufP1, degp, bp1, bufP2, NP);
  scatter_edges<<<cdiv(EP, 8), 256, 0, stream>>>(bufP1, ei_p, degp, bufP2, EP);
  gemm_bf16    <<<cdiv(tilesP, 4), 128, 0, stream>>>(bufP2, Wp2p, bufP1, H, tilesP);
  scatter_init <<<cdiv(NP * H, 256), 256, 0, stream>>>(bufP1, degp, bp2, bufP2, NP);
  scatter_edges<<<cdiv(EP, 8), 256, 0, stream>>>(bufP1, ei_p, degp, bufP2, EP);
  gemm_bf16    <<<cdiv(tilesP, 4), 128, 0, stream>>>(bufP2, Wftop, bufP1, H, tilesP);  // hp

  // 5) substrate GCN
  gemm_bf16    <<<cdiv(tilesS, 4), 128, 0, stream>>>(x_s, Ws1p, bufS1, SD, tilesS);
  scatter_init <<<cdiv(NS * H, 256), 256, 0, stream>>>(bufS1, degs, bs1, bufS2, NS);
  scatter_edges<<<cdiv(ES, 8), 256, 0, stream>>>(bufS1, ei_s, degs, bufS2, ES);
  gemm_bf16    <<<cdiv(tilesS, 4), 128, 0, stream>>>(bufS2, Ws2p, bufS1, H, tilesS);
  scatter_init <<<cdiv(NS * H, 256), 256, 0, stream>>>(bufS1, degs, bs2, bufS2, NS);
  scatter_edges<<<cdiv(ES, 8), 256, 0, stream>>>(bufS1, ei_s, degs, bufS2, ES);
  gemm_bf16    <<<cdiv(tilesS, 4), 128, 0, stream>>>(bufS2, Wfbot, bufS1, H, tilesS);  // hs

  // 6) per-edge link prediction (gather + ReLU-dot, L2-resident tables)
  link_pred<<<cdiv(EL, 8), 256, 0, stream>>>(bufP1, bufS1, ed_p, ed_s,
                                             b1fold, Wl2, bl2, out, EL);
}
